// TorchMD_T_2233382993956
// MI455X (gfx1250) — matrix operand order repaired and checked by reading, not compiled, not run on hardware
//
#include <hip/hip_runtime.h>
#include <hip/hip_bf16.h>

// TorchMD attention layer for gfx1250 (MI455X).
// H=256 hidden, R=64 rbf, N=20000 nodes, E=320000 edges.
// All GEMMs via v_wmma_f32_16x16x32_f16 (f32 accumulate). Edge filters dk/dv are
// computed tile-wise in LDS and never written to HBM. q/k/v kept f16 (10MB each)
// so the random edge gathers and the agg atomics live in the 192MB L2.
// edge_attr tiles staged into LDS via GLOBAL_LOAD_ASYNC_TO_LDS (ASYNCcnt path).

#define Hh 256
#define Rr 64
#define NNODE 20000
#define NEDGE 320000

typedef __attribute__((ext_vector_type(16))) _Float16 v16h;
typedef __attribute__((ext_vector_type(8)))  float    v8f;
typedef int v4i __attribute__((vector_size(16)));

__device__ __forceinline__ float silu_f(float x) { return x / (1.0f + __expf(-x)); }

// ---- WMMA fragment loaders (ISA 7.12.2, wave32, 16x16x32 f16) ----
// A (16x32, row-major src, leading dim lda):
//   lanes 0-15 : M=lane,    halves [0:8)=K 0..7,  [8:16)=K 16..23
//   lanes 16-31: M=lane-16, halves [0:8)=K 8..15, [8:16)=K 24..31
__device__ __forceinline__ v16h load_a_frag(const _Float16* A, int lda) {
  const int lane = threadIdx.x & 31;
  const _Float16* p = A + (size_t)(lane & 15) * lda + ((lane < 16) ? 0 : 8);
  v16h a;
#pragma unroll
  for (int i = 0; i < 8; ++i) { a[i] = p[i]; a[8 + i] = p[16 + i]; }
  return a;
}
// Same layout, but source is f32 (LDS-staged edge_attr); convert at load.
__device__ __forceinline__ v16h load_a_frag_f32(const float* A, int lda) {
  const int lane = threadIdx.x & 31;
  const float* p = A + (size_t)(lane & 15) * lda + ((lane < 16) ? 0 : 8);
  v16h a;
#pragma unroll
  for (int i = 0; i < 8; ++i) { a[i] = (_Float16)p[i]; a[8 + i] = (_Float16)p[16 + i]; }
  return a;
}
// B (32x16). Source is W row-major [outCh][K] (out = x @ W^T), so B(k,n)=W[n][k]:
//   lane n (0-15): halves [0:8)=K 0..7, [8:16)=K 16..23 of column n; lane n+16: K 8..15, 24..31.
__device__ __forceinline__ v16h load_b_frag(const _Float16* W, int ldw) {
  const int lane = threadIdx.x & 31;
  const _Float16* p = W + (size_t)(lane & 15) * ldw + ((lane < 16) ? 0 : 8);
  v16h b;
#pragma unroll
  for (int i = 0; i < 8; ++i) { b[i] = p[i]; b[8 + i] = p[16 + i]; }
  return b;
}

// ---- 1. LayerNorm -> f16, one wave per row ----
__global__ __launch_bounds__(256) void ln_kernel(const float* __restrict__ x,
                                                 const float* __restrict__ g,
                                                 const float* __restrict__ b,
                                                 _Float16* __restrict__ xh) {
  const int wave = threadIdx.x >> 5, lane = threadIdx.x & 31;
  const int row = blockIdx.x * 8 + wave;
  const float* xr = x + (size_t)row * Hh;
  float v[8], s = 0.f, ss = 0.f;
#pragma unroll
  for (int i = 0; i < 8; ++i) { float t = xr[lane * 8 + i]; v[i] = t; s += t; ss += t * t; }
#pragma unroll
  for (int off = 16; off; off >>= 1) { s += __shfl_xor(s, off, 32); ss += __shfl_xor(ss, off, 32); }
  const float mu = s * (1.0f / Hh);
  const float var = ss * (1.0f / Hh) - mu * mu;
  const float rs = rsqrtf(var + 1e-5f);
  _Float16* o = xh + (size_t)row * Hh;
#pragma unroll
  for (int i = 0; i < 8; ++i) {
    const int h = lane * 8 + i;
    o[h] = (_Float16)((v[i] - mu) * rs * g[h] + b[h]);
  }
}

// ---- helpers: cast / zero ----
__global__ void cast_f2h(const float* __restrict__ in, _Float16* __restrict__ out, int n) {
  const int i = blockIdx.x * blockDim.x + threadIdx.x;
  if (i < n) out[i] = (_Float16)in[i];
}
__global__ void zero_f32(float* __restrict__ p, int n) {
  const int i = blockIdx.x * blockDim.x + threadIdx.x;
  if (i < n) p[i] = 0.0f;
}

// ---- 2. QKV GEMM: one wave per 16x64 tile strip (4 col tiles share A-frags) ----
// waves: 1250 m-tiles * 3 matrices * 4 col-groups = 15000 -> 1875 blocks.
__global__ __launch_bounds__(256) void qkv_kernel(
    const _Float16* __restrict__ xh,
    const _Float16* __restrict__ Wq, const float* __restrict__ bq,
    const _Float16* __restrict__ Wk, const float* __restrict__ bk,
    const _Float16* __restrict__ Wv, const float* __restrict__ bv,
    _Float16* __restrict__ q, _Float16* __restrict__ k, _Float16* __restrict__ v) {
  const int wave = threadIdx.x >> 5, lane = threadIdx.x & 31;
  const int gw = blockIdx.x * 8 + wave;        // 0..14999
  const int mt = gw / 12;
  const int j = gw % 12;
  const int mat = j >> 2, ng = j & 3;
  const _Float16* W = (mat == 0) ? Wq : ((mat == 1) ? Wk : Wv);
  const float* bias = (mat == 0) ? bq : ((mat == 1) ? bk : bv);
  _Float16* out = (mat == 0) ? q : ((mat == 1) ? k : v);
  const int m0 = mt * 16, n0 = ng * 64;
  v8f c[4] = {};
#pragma unroll
  for (int ks = 0; ks < 8; ++ks) {
    const v16h a = load_a_frag(xh + (size_t)m0 * Hh + ks * 32, Hh);
#pragma unroll
    for (int u = 0; u < 4; ++u) {
      const v16h b = load_b_frag(W + (size_t)(n0 + u * 16) * Hh + ks * 32, Hh);
      c[u] = __builtin_amdgcn_wmma_f32_16x16x32_f16(false, a, false, b, (short)0, c[u],
                                                    false, false);
    }
  }
  const int rb = m0 + ((lane >= 16) ? 8 : 0);
#pragma unroll
  for (int u = 0; u < 4; ++u) {
    const int col = n0 + u * 16 + (lane & 15);
    const float bb = bias[col];
#pragma unroll
    for (int i = 0; i < 8; ++i)
      out[(size_t)(rb + i) * Hh + col] = (_Float16)(c[u][i] + bb);
  }
}

// ---- 3. Edge kernel: 16 edges/block. dk/dv via WMMA into LDS, attention, scatter ----
__global__ __launch_bounds__(256) void edge_kernel(
    const float* __restrict__ ea, const int* __restrict__ ei,
    const float* __restrict__ ew,
    const _Float16* __restrict__ Wdk, const float* __restrict__ bdk,
    const _Float16* __restrict__ Wdv, const float* __restrict__ bdv,
    const _Float16* __restrict__ qh, const _Float16* __restrict__ kh,
    const _Float16* __restrict__ vh, float* __restrict__ agg) {
  __shared__ float eaF[16 * Rr];         // edge_attr tile (f32, async-staged)
  __shared__ float dkS[16][Hh];          // silu(edge_attr@Wdk^T + b)
  __shared__ float dvS[16][Hh];
  __shared__ float attnS[16];
  __shared__ float cwS[16];
  __shared__ int srcS[16], dstS[16];

  const int t = threadIdx.x;
  const int e0 = blockIdx.x * 16;

  // stage edge_attr [16x64] f32 -> LDS: one async b128 per thread (ASYNCcnt)
  {
    const float* gsrc = ea + (size_t)e0 * Rr + t * 4;
    float* ldst = eaF + t * 4;
#if __has_builtin(__builtin_amdgcn_global_load_async_to_lds_b128)
    __builtin_amdgcn_global_load_async_to_lds_b128(
        (__attribute__((address_space(1))) v4i*)gsrc,
        (__attribute__((address_space(3))) v4i*)ldst, 0, 0);
#if __has_builtin(__builtin_amdgcn_s_wait_asynccnt)
    __builtin_amdgcn_s_wait_asynccnt(0);
#else
    asm volatile("s_wait_asynccnt 0" ::: "memory");
#endif
#else
    *(float4*)ldst = *(const float4*)gsrc;
#endif
  }
  if (t < 16) {
    srcS[t] = ei[e0 + t];
    dstS[t] = ei[NEDGE + e0 + t];
    const float r = ew[e0 + t];
    const float cc = 0.5f * (__cosf(r * 0.62831853071795864769f) + 1.0f);  // pi/5
    cwS[t] = (r < 5.0f) ? cc : 0.0f;
  }
  __syncthreads();

  const int wave = t >> 5, lane = t & 31;
  // A-frags (the 16x64 edge tile) are identical for every tile this wave computes:
  const v16h a0 = load_a_frag_f32(eaF, Rr);        // K 0..31
  const v16h a1 = load_a_frag_f32(eaF + 32, Rr);   // K 32..63
  // dk/dv GEMM: 32 col-tiles (16 per matrix), 4 per wave
#pragma unroll
  for (int tt = 0; tt < 4; ++tt) {
    const int j = wave * 4 + tt;             // 0..31
    const int mat = j >> 4, nt = j & 15, n0 = nt * 16;
    const _Float16* W = mat ? Wdv : Wdk;
    const float* bias = mat ? bdv : bdk;
    v8f c = {};
    const v16h b0 = load_b_frag(W + (size_t)n0 * Rr, Rr);
    c = __builtin_amdgcn_wmma_f32_16x16x32_f16(false, a0, false, b0, (short)0, c, false, false);
    const v16h b1 = load_b_frag(W + (size_t)n0 * Rr + 32, Rr);
    c = __builtin_amdgcn_wmma_f32_16x16x32_f16(false, a1, false, b1, (short)0, c, false, false);
    const int col = n0 + (lane & 15);
    const float bb = bias[col];
    const int rb = (lane >= 16) ? 8 : 0;
    float(*dstL)[Hh] = mat ? dvS : dkS;
#pragma unroll
    for (int i = 0; i < 8; ++i) dstL[rb + i][col] = silu_f(c[i] + bb);
  }
  __syncthreads();

  // attention: attn[e] = silu( sum_h q[dst]*k[src]*dk ) * cutoff; 2 edges per wave
#pragma unroll
  for (int eei = 0; eei < 2; ++eei) {
    const int e = wave * 2 + eei;
    const _Float16* qr = qh + (size_t)dstS[e] * Hh;
    const _Float16* kr = kh + (size_t)srcS[e] * Hh;
    float acc = 0.0f;
#pragma unroll
    for (int ii = 0; ii < 8; ++ii) {
      const int h = lane + 32 * ii;
      acc += (float)qr[h] * (float)kr[h] * dkS[e][h];
    }
#pragma unroll
    for (int off = 16; off; off >>= 1) acc += __shfl_xor(acc, off, 32);
    if (lane == 0) attnS[e] = silu_f(acc) * cwS[e];
  }
  __syncthreads();

  // msg = v[src]*dv*attn, scatter-add into agg[dst] (L2-resident atomics)
  const int e = t >> 4;
  const int hb = t & 15;
  const _Float16* vr = vh + (size_t)srcS[e] * Hh;
  float* ar = agg + (size_t)dstS[e] * Hh;
  const float at = attnS[e];
#pragma unroll
  for (int ii = 0; ii < 16; ++ii) {
    const int h = hb + 16 * ii;
    atomicAdd(ar + h, (float)vr[h] * dvS[e][h] * at);
  }
}

// ---- 4. out = x + agg @ Wo^T + bo (one wave per 16x64 strip) ----
// waves: 1250 m-tiles * 4 col-groups = 5000 -> 625 blocks.
__global__ __launch_bounds__(256) void out_kernel(
    const _Float16* __restrict__ ah, const _Float16* __restrict__ Wo,
    const float* __restrict__ bo, const float* __restrict__ x,
    float* __restrict__ out) {
  const int wave = threadIdx.x >> 5, lane = threadIdx.x & 31;
  const int gw = blockIdx.x * 8 + wave;  // 0..4999
  const int mt = gw >> 2, ng = gw & 3;
  const int m0 = mt * 16, n0 = ng * 64;
  v8f c[4] = {};
#pragma unroll
  for (int ks = 0; ks < 8; ++ks) {
    const v16h a = load_a_frag(ah + (size_t)m0 * Hh + ks * 32, Hh);
#pragma unroll
    for (int u = 0; u < 4; ++u) {
      const v16h b = load_b_frag(Wo + (size_t)(n0 + u * 16) * Hh + ks * 32, Hh);
      c[u] = __builtin_amdgcn_wmma_f32_16x16x32_f16(false, a, false, b, (short)0, c[u],
                                                    false, false);
    }
  }
  const int rb = m0 + ((lane >= 16) ? 8 : 0);
#pragma unroll
  for (int u = 0; u < 4; ++u) {
    const int col = n0 + u * 16 + (lane & 15);
    const float bb = bo[col];
#pragma unroll
    for (int i = 0; i < 8; ++i) {
      const size_t idx = (size_t)(rb + i) * Hh + col;
      out[idx] = x[idx] + c[u][i] + bb;
    }
  }
}

extern "C" void kernel_launch(void* const* d_in, const int* in_sizes, int n_in,
                              void* d_out, int out_size, void* d_ws, size_t ws_size,
                              hipStream_t stream) {
  (void)in_sizes; (void)n_in; (void)out_size; (void)ws_size;
  const float* x   = (const float*)d_in[0];
  const int*   ei  = (const int*)  d_in[1];   // JAX default x64-off => int32
  const float* ew  = (const float*)d_in[2];
  const float* ea  = (const float*)d_in[3];
  const float* lng = (const float*)d_in[4];
  const float* lnb = (const float*)d_in[5];
  const float* Wq  = (const float*)d_in[6];  const float* bq  = (const float*)d_in[7];
  const float* Wk  = (const float*)d_in[8];  const float* bk  = (const float*)d_in[9];
  const float* Wv  = (const float*)d_in[10]; const float* bv  = (const float*)d_in[11];
  const float* Wo  = (const float*)d_in[12]; const float* bo  = (const float*)d_in[13];
  const float* Wdk = (const float*)d_in[14]; const float* bdk = (const float*)d_in[15];
  const float* Wdv = (const float*)d_in[16]; const float* bdv = (const float*)d_in[17];
  float* out = (float*)d_out;

  // workspace carve-out (~62 MB total)
  char* ws = (char*)d_ws;
  size_t off = 0;
  auto carve = [&](size_t bytes) -> void* {
    void* p = ws + off;
    off += (bytes + 255) & ~(size_t)255;
    return p;
  };
  const size_t NH = (size_t)NNODE * Hh;  // 5,120,000
  _Float16* xh   = (_Float16*)carve(NH * 2);   // x_norm f16; reused as agg_h later
  _Float16* qh   = (_Float16*)carve(NH * 2);
  _Float16* kh   = (_Float16*)carve(NH * 2);
  _Float16* vh   = (_Float16*)carve(NH * 2);
  float*    agg  = (float*)   carve(NH * 4);
  _Float16* Wqh  = (_Float16*)carve((size_t)Hh * Hh * 2);
  _Float16* Wkh  = (_Float16*)carve((size_t)Hh * Hh * 2);
  _Float16* Wvh  = (_Float16*)carve((size_t)Hh * Hh * 2);
  _Float16* Woh  = (_Float16*)carve((size_t)Hh * Hh * 2);
  _Float16* Wdkh = (_Float16*)carve((size_t)Hh * Rr * 2);
  _Float16* Wdvh = (_Float16*)carve((size_t)Hh * Rr * 2);

  // 1. layernorm -> f16
  ln_kernel<<<NNODE / 8, 256, 0, stream>>>(x, lng, lnb, xh);
  // 2. cast weights -> f16
  const int HH = Hh * Hh, HR = Hh * Rr;
  cast_f2h<<<(HH + 255) / 256, 256, 0, stream>>>(Wq, Wqh, HH);
  cast_f2h<<<(HH + 255) / 256, 256, 0, stream>>>(Wk, Wkh, HH);
  cast_f2h<<<(HH + 255) / 256, 256, 0, stream>>>(Wv, Wvh, HH);
  cast_f2h<<<(HH + 255) / 256, 256, 0, stream>>>(Wo, Woh, HH);
  cast_f2h<<<(HR + 255) / 256, 256, 0, stream>>>(Wdk, Wdkh, HR);
  cast_f2h<<<(HR + 255) / 256, 256, 0, stream>>>(Wdv, Wdvh, HR);
  // 3. zero accumulator (ws is poisoned between runs)
  zero_f32<<<(int)((NH + 255) / 256), 256, 0, stream>>>(agg, (int)NH);
  // 4. q/k/v GEMMs: 15000 waves / 8 = 1875 blocks
  qkv_kernel<<<1875, 256, 0, stream>>>(xh, Wqh, bq, Wkh, bk, Wvh, bv, qh, kh, vh);
  // 5. edge filters + attention + scatter: 20000 blocks of 16 edges
  edge_kernel<<<NEDGE / 16, 256, 0, stream>>>(ea, ei, ew, Wdkh, bdk, Wdvh, bdv,
                                              qh, kh, vh, agg);
  // 6. agg -> f16 (reuse xh region; x_norm is dead now)
  cast_f2h<<<(int)((NH + 255) / 256), 256, 0, stream>>>(agg, xh, (int)NH);
  // 7. residual output GEMM: 5000 waves / 8 = 625 blocks
  out_kernel<<<625, 256, 0, stream>>>(xh, Woh, bo, x, out);
}
